// Adex_80693845557792
// MI455X (gfx1250) — matrix operand order repaired and checked by reading, not compiled, not run on hardware
//
#include <hip/hip_runtime.h>

// ---------------- problem constants ----------------
constexpr int BATCH = 8192;
constexpr int N_IN  = 256;
constexpr int UNITS = 1024;

constexpr float kThr    = -50.4f;
constexpr float kEl     = -70.6f;
constexpr float kC      = 281.0f;
constexpr float kDeltaT = 2.0f;
constexpr float kVReset = -70.6f;
constexpr float kTauW   = 144.0f;
constexpr float kA      = 4.0f;
constexpr float kB      = 0.0805f;
constexpr float kDt     = 1.0f;
constexpr float kDtGlC  = kDt * 30.0f / kC;        // DT*GL/C
constexpr float kDtATw  = kDt * kA / kTauW;        // DT*A/TAUW
constexpr float kExpCap = 30.0f / kDtGlC;          // = 281.0
constexpr int   kNRef   = 5;

// ---------------- tiling ----------------
constexpr int BM = 128, BN = 128, BK = 32;
constexpr int LDT = BK + 8;                        // padded LDS stride (bf16 elems)
constexpr int KSTEPS = (N_IN + UNITS) / BK;        // 40
constexpr int KSPLIT = N_IN / BK;                  // 8 (input-phase steps)

typedef __attribute__((ext_vector_type(16))) __bf16 bf16x16;
typedef __attribute__((ext_vector_type(8)))  float  f32x8;

union Frag16 { bf16x16 v; uint4 u[2]; };
union F4     { float4 v; float e[4]; };

// pack two f32 -> two bf16 in one dword: round-half-up then v_perm_b32 grabs
// the high halves of both floats. 3 VALU ops per packed dword.
static __device__ __forceinline__ unsigned pk_bf16(float lo, float hi) {
  unsigned a = __float_as_uint(lo) + 0x8000u;
  unsigned b = __float_as_uint(hi) + 0x8000u;
  // selector bytes (LSB first): 2,3 -> S1[31:16] ; 6,7 -> S0[31:16]
  return __builtin_amdgcn_perm(b, a, 0x07060302u);
}

__global__ __launch_bounds__(256)
void adex_fused_kernel(const float* __restrict__ inputs,   // [BATCH, N_IN]
                       const float* __restrict__ oldV,     // [BATCH, UNITS]
                       const int*   __restrict__ oldR,     // [BATCH, UNITS]
                       const float* __restrict__ oldW,     // [BATCH, UNITS]
                       const float* __restrict__ oldZ,     // [BATCH, UNITS]
                       const float* __restrict__ wIn,      // [N_IN, UNITS]
                       const float* __restrict__ wRec,     // [UNITS, UNITS]
                       float* __restrict__ outV,
                       float* __restrict__ outZ,
                       int*   __restrict__ outR,
                       float* __restrict__ outW) {
  // double-buffered tiles: A row-major [m][k], B transposed [n][k], bf16
  __shared__ __align__(16) unsigned short sA[2][BM * LDT];
  __shared__ __align__(16) unsigned short sB[2][BN * LDT];

  const int tid    = threadIdx.x;
  const int lane   = tid & 31;
  const int wave   = tid >> 5;
  const int waveM  = wave & 3;          // 4 waves along M
  const int waveN  = wave >> 2;         // 2 waves along N
  const int h      = lane >> 4;         // lane half
  const int l16    = lane & 15;

  const int blockM = blockIdx.y * BM;
  const int blockN = blockIdx.x * BN;

  // A staging: thread owns 16 contiguous floats of one row
  const int aRow  = tid >> 1;                 // 0..127
  const int aCol  = (tid & 1) * 16;           // 0 or 16
  // B staging: thread owns a 4(K) x 4(N) block -> register transpose
  const int bKGrp = tid >> 5;                 // 0..7  -> K rows bKGrp*4..+3
  const int bNCol = (tid & 31) * 4;           // 0..124

  F4 aReg[4], bReg[4];

  auto loadA = [&](int ks) {
    const float* src = (ks < KSPLIT)
        ? inputs + (size_t)(blockM + aRow) * N_IN  + ks * BK + aCol
        : oldZ   + (size_t)(blockM + aRow) * UNITS + (ks * BK - N_IN) + aCol;
#pragma unroll
    for (int j = 0; j < 4; ++j)
      aReg[j].v = *reinterpret_cast<const float4*>(src + 4 * j);
  };

  auto loadB = [&](int ks) {
#pragma unroll
    for (int j = 0; j < 4; ++j) {
      const int kr = ks * BK + bKGrp * 4 + j;
      const float* src = (ks < KSPLIT)
          ? wIn  + (size_t)kr * UNITS + blockN + bNCol
          : wRec + (size_t)(kr - N_IN) * UNITS + blockN + bNCol;
      bReg[j].v = *reinterpret_cast<const float4*>(src);
    }
  };

  auto storeA = [&](int buf) {
#pragma unroll
    for (int j = 0; j < 4; ++j) {
      uint2 p;
      p.x = pk_bf16(aReg[j].e[0], aReg[j].e[1]);
      p.y = pk_bf16(aReg[j].e[2], aReg[j].e[3]);
      *reinterpret_cast<uint2*>(&sA[buf][aRow * LDT + aCol + 4 * j]) = p;
    }
  };

  auto storeB = [&](int ks, int buf) {
    // no-autapse diagonal: uniform (scalar) test whether this k-slab touches it
    const int k0r = ks * BK - N_IN;
    if (ks >= KSPLIT && k0r < blockN + BN && k0r + BK > blockN) {
#pragma unroll
      for (int j = 0; j < 4; ++j) {
        const int gk = k0r + bKGrp * 4 + j;
#pragma unroll
        for (int i = 0; i < 4; ++i)
          if (blockN + bNCol + i == gk) bReg[j].e[i] = 0.0f;
      }
    }
#pragma unroll
    for (int i = 0; i < 4; ++i) {      // register transpose: column i -> [n][k]
      uint2 p;
      p.x = pk_bf16(bReg[0].e[i], bReg[1].e[i]);
      p.y = pk_bf16(bReg[2].e[i], bReg[3].e[i]);
      *reinterpret_cast<uint2*>(&sB[buf][(bNCol + i) * LDT + bKGrp * 4]) = p;
    }
  };

  f32x8 acc[2][4];
  const f32x8 vzero = {0.f,0.f,0.f,0.f,0.f,0.f,0.f,0.f};
#pragma unroll
  for (int i = 0; i < 2; ++i)
#pragma unroll
    for (int j = 0; j < 4; ++j) acc[i][j] = vzero;

  // prologue: stage tile 0
  loadA(0); loadB(0);
  storeA(0); storeB(0, 0);
  __syncthreads();

  for (int ks = 0; ks < KSTEPS; ++ks) {
    const int cur = ks & 1;
    const int nxt = cur ^ 1;

    if (ks + 1 < KSTEPS) { loadA(ks + 1); loadB(ks + 1); }   // prefetch to regs

    // ---- fragments + WMMA on current buffer ----
    Frag16 fa[2], fb[4];
#pragma unroll
    for (int tm = 0; tm < 2; ++tm) {
      const int row = waveM * 32 + tm * 16 + l16;
      const uint4* p = reinterpret_cast<const uint4*>(&sA[cur][row * LDT + h * 8]);
      fa[tm].u[0] = p[0];     // K = h*8 .. h*8+7
      fa[tm].u[1] = p[2];     // K = 16+h*8 .. 16+h*8+7
    }
#pragma unroll
    for (int tn = 0; tn < 4; ++tn) {
      const int n = waveN * 64 + tn * 16 + l16;
      const uint4* q = reinterpret_cast<const uint4*>(&sB[cur][n * LDT + h * 8]);
      fb[tn].u[0] = q[0];     // K = h*8 .. h*8+7
      fb[tn].u[1] = q[2];     // K = 16+h*8 .. 16+h*8+7
    }
#pragma unroll
    for (int tm = 0; tm < 2; ++tm)
#pragma unroll
      for (int tn = 0; tn < 4; ++tn)
        acc[tm][tn] = __builtin_amdgcn_wmma_f32_16x16x32_bf16(
            false, fa[tm].v, false, fb[tn].v, (short)0, acc[tm][tn], false, false);

    if (ks + 1 < KSTEPS) { storeA(nxt); storeB(ks + 1, nxt); }  // fill other buffer
    __syncthreads();
  }

  // ---- epilogue: AdEx elementwise state update (streamed, non-temporal) ----
#pragma unroll
  for (int tm = 0; tm < 2; ++tm) {
#pragma unroll
    for (int tn = 0; tn < 4; ++tn) {
#pragma unroll
      for (int i = 0; i < 8; ++i) {
        const int row = blockM + waveM * 32 + tm * 16 + h * 8 + i;
        const int col = blockN + waveN * 64 + tn * 16 + l16;
        const size_t idx = (size_t)row * UNITS + col;

        const float it  = acc[tm][tn][i];
        const float ov  = __builtin_nontemporal_load(oldV + idx);
        const float ow  = __builtin_nontemporal_load(oldW + idx);
        const int   orf = __builtin_nontemporal_load(oldR + idx);
        const float oz  = oldZ[idx];   // also read by GEMMs of other blocks: keep cached

        float ex = __expf((ov - kThr) * (1.0f / kDeltaT));
        ex = fminf(ex, kExpCap);                       // upper clip (lower moot: exp>0)

        float nv = ov - kDtGlC * (ov - kEl)
                 + (kDtGlC * kDeltaT) * ex
                 + (it - ow) * (kDt / kC);
        nv = (oz > 0.5f) ? kVReset : nv;

        const float nw = ow - (kDt / kTauW) * ow + kDtATw * (ov - kEl) + kB * oz;

        const float vs = (nv - kThr) * (1.0f / (kThr - kEl));
        float nz = (vs > 0.0f) ? 1.0f : 0.0f;
        if (orf > 0) nz = 0.0f;

        int nr = orf - 1 + (int)(nz * (float)kNRef);
        nr = nr < 0 ? 0 : (nr > kNRef ? kNRef : nr);

        __builtin_nontemporal_store(nv, outV + idx);
        __builtin_nontemporal_store(nz, outZ + idx);
        __builtin_nontemporal_store(nr, outR + idx);
        __builtin_nontemporal_store(nw, outW + idx);
      }
    }
  }
}

extern "C" void kernel_launch(void* const* d_in, const int* in_sizes, int n_in,
                              void* d_out, int out_size, void* d_ws, size_t ws_size,
                              hipStream_t stream) {
  const float* inputs = (const float*)d_in[0];
  const float* old_v  = (const float*)d_in[1];
  const int*   old_r  = (const int*)  d_in[2];
  const float* old_w  = (const float*)d_in[3];
  const float* old_z  = (const float*)d_in[4];
  const float* w_in   = (const float*)d_in[5];
  const float* w_rec  = (const float*)d_in[6];

  float* out = (float*)d_out;
  const size_t N = (size_t)BATCH * UNITS;

  dim3 grid(UNITS / BN, BATCH / BM);   // (8, 64)
  dim3 block(256);
  adex_fused_kernel<<<grid, block, 0, stream>>>(
      inputs, old_v, old_r, old_w, old_z, w_in, w_rec,
      /*new_v*/ out,
      /*new_z*/ out + N,
      /*new_r*/ (int*)(out + 2 * N),
      /*new_w*/ out + 3 * N);
}
